// MultiHeadedAttention_23854248362592
// MI455X (gfx1250) — compile-verified, matrix operand order checked by reading, not run on hardware
//
#include <hip/hip_runtime.h>
#include <hip/hip_bf16.h>

// ---------------- problem constants ----------------
#define kB   4
#define kS   2048
#define kH   16
#define kD   1024
#define kDK  64

typedef __bf16 bf16;
typedef __attribute__((ext_vector_type(16))) __bf16 v16bf;
typedef __attribute__((ext_vector_type(8)))  __bf16 v8bf;
typedef __attribute__((ext_vector_type(8)))  float  v8f;
typedef __attribute__((ext_vector_type(4))) unsigned int u32x4;
typedef __attribute__((ext_vector_type(8))) int          i32x8;
typedef __attribute__((ext_vector_type(4))) int          i32x4;

// group1 word0: data_size=2B (1<<16) | pad_enable (1<<20) | pad_interval<<22 | pad_amount<<25
// pitch-72 tile rows: 128B data rows (32 DW -> interval code 4) + 16B pad (4 DW -> amount code 3)
#define TDM_G1W0_P72  ((1u << 16) | (1u << 20) | (4u << 22) | (3u << 25))
// pitch-40 tile rows: 64B data rows (16 DW -> interval code 3) + 16B pad (4 DW -> amount code 3)
#define TDM_G1W0_P40  ((1u << 16) | (1u << 20) | (3u << 22) | (3u << 25))

// Issue a 2-D TDM tile load: tile (tile0 x tile1) of 2-byte elements from a
// row-major tensor (line length dim0, row stride stride0, dim1 rows) into LDS
// at byte offset lds_off, with LDS row padding per g1w0.  D# packing per
// CDNA5 ISA (groups 2-3 zero => 2-D tensor).
__device__ __forceinline__ void tdm_load_2d(
    const void* gaddr, unsigned lds_off, unsigned g1w0,
    unsigned dim0, unsigned dim1, unsigned tile0, unsigned tile1,
    unsigned stride0)
{
    unsigned long long ga = (unsigned long long)(uintptr_t)gaddr;
    u32x4 g0;
    g0[0] = 1u;                                            // count=1, no gather
    g0[1] = lds_off;                                       // lds_addr
    g0[2] = (unsigned)(ga & 0xffffffffu);                  // global_addr[31:0]
    g0[3] = (unsigned)((ga >> 32) & 0x01ffffffu)           // global_addr[56:32]
          | 0x80000000u;                                   // type=2 ("image")
    i32x8 g1;
    g1[0] = (int)g1w0;
    g1[1] = (int)((dim0 & 0xffffu) << 16);                 // tensor_dim0[15:0]
    g1[2] = (int)((dim0 >> 16) | ((dim1 & 0xffffu) << 16));// dim0[31:16], dim1[15:0]
    g1[3] = (int)((dim1 >> 16) | (tile0 << 16));           // dim1[31:16], tile_dim0
    g1[4] = (int)(tile1 & 0xffffu);                        // tile_dim1 (tile_dim2=0)
    g1[5] = (int)stride0;                                  // tensor_dim0_stride[31:0]
    g1[6] = 0;                                             // stride0 hi, stride1 lo
    g1[7] = 0;
    i32x4 z4 = {0, 0, 0, 0};
    i32x8 z8 = {0, 0, 0, 0, 0, 0, 0, 0};
    // clang-23 / therock form: (g0, g1, g2, g3, extra, cpol)
    __builtin_amdgcn_tensor_load_to_lds(g0, g1, z4, z4, z8, 0);
}

// Build a 16-bit A/B fragment (16x32 tile, NT form) from two 16-byte chunks.
__device__ __forceinline__ v16bf frag_nt(const bf16* p) {
    v8bf lo = *reinterpret_cast<const v8bf*>(p);
    v8bf hi = *reinterpret_cast<const v8bf*>(p + 16);
    return __builtin_shufflevector(lo, hi, 0,1,2,3,4,5,6,7,8,9,10,11,12,13,14,15);
}

__device__ __forceinline__ v8f wmma_bf16(v16bf a, v16bf b, v8f c) {
    return __builtin_amdgcn_wmma_f32_16x16x32_bf16(
        false, a, false, b, (short)0, c, false, false);
}

// ---------------- f32 -> bf16 conversion ----------------
__global__ void cvt_f32_bf16(const float* __restrict__ in, bf16* __restrict__ out, int n) {
    int i = (blockIdx.x * blockDim.x + threadIdx.x) * 8;
    int stride = gridDim.x * blockDim.x * 8;
    for (; i < n; i += stride) {
        float4 a = *reinterpret_cast<const float4*>(in + i);
        float4 b = *reinterpret_cast<const float4*>(in + i + 4);
        v8bf o;
        o[0] = (bf16)a.x; o[1] = (bf16)a.y; o[2] = (bf16)a.z; o[3] = (bf16)a.w;
        o[4] = (bf16)b.x; o[5] = (bf16)b.y; o[6] = (bf16)b.z; o[7] = (bf16)b.w;
        *reinterpret_cast<v8bf*>(out + i) = o;
    }
}

// ---------------- NT GEMM: Y = (X @ W^T + bias) * scale ----------------
// X: [M,K] bf16 row-major; W: [N,K] bf16 row-major (B = W^T is K-contig).
// Block: 256 thr = 8 waves (2x4); wave tile 32x32; block tile 64x128.
// Tiles staged by the Tensor Data Mover (wave0: A, wave1: B).
__global__ __launch_bounds__(256) void gemm_nt_bf16(
    const bf16* __restrict__ A, const bf16* __restrict__ W,
    const float* __restrict__ bias, bf16* __restrict__ outb,
    float* __restrict__ outf, int M, int N, int K, float scale)
{
    __shared__ bf16 As[64][40];    // TDM-padded pitch 40 elems = 80B
    __shared__ bf16 Bs[128][40];

    const int tid  = threadIdx.x;
    const int lane = tid & 31;
    const int wave = tid >> 5;
    const int wm   = wave >> 2;
    const int wn   = wave & 3;
    const int half = lane >> 4;
    const int ln   = lane & 15;
    const int bm   = blockIdx.x * 64;
    const int bn   = blockIdx.y * 128;

    const unsigned As_off = (unsigned)(uintptr_t)&As[0][0];
    const unsigned Bs_off = (unsigned)(uintptr_t)&Bs[0][0];

    v8f c[2][2] = {};

    for (int k0 = 0; k0 < K; k0 += 32) {
        // TDM-stage the A (64x32) and B (128x32) tiles into padded LDS rows
        if (wave == 0) {
            tdm_load_2d(A + (size_t)bm * K + k0, As_off, TDM_G1W0_P40,
                        (unsigned)K, (unsigned)M, 32u, 64u, (unsigned)K);
            __builtin_amdgcn_s_wait_tensorcnt(0);
        } else if (wave == 1) {
            tdm_load_2d(W + (size_t)bn * K + k0, Bs_off, TDM_G1W0_P40,
                        (unsigned)K, (unsigned)N, 32u, 128u, (unsigned)K);
            __builtin_amdgcn_s_wait_tensorcnt(0);
        }
        __syncthreads();

        const int kb = 8 * half;
        v16bf a0 = frag_nt(&As[wm * 32 +      ln][kb]);
        v16bf a1 = frag_nt(&As[wm * 32 + 16 + ln][kb]);
        v16bf b0 = frag_nt(&Bs[wn * 32 +      ln][kb]);
        v16bf b1 = frag_nt(&Bs[wn * 32 + 16 + ln][kb]);
        c[0][0] = wmma_bf16(a0, b0, c[0][0]);
        c[0][1] = wmma_bf16(a0, b1, c[0][1]);
        c[1][0] = wmma_bf16(a1, b0, c[1][0]);
        c[1][1] = wmma_bf16(a1, b1, c[1][1]);
        __syncthreads();
    }

    // epilogue: C layout — VGPR r holds row (r + 8*half), col = ln
    for (int i = 0; i < 2; ++i)
        for (int j = 0; j < 2; ++j) {
            int colg = bn + wn * 32 + j * 16 + ln;
            float bv = bias ? bias[colg] : 0.0f;
            for (int r = 0; r < 8; ++r) {
                int rowg = bm + wm * 32 + i * 16 + r + 8 * half;
                float val = (c[i][j][r] + bv) * scale;
                if (outf) outf[(size_t)rowg * N + colg] = val;
                else      outb[(size_t)rowg * N + colg] = (bf16)val;
            }
        }
}

// ---------------- flash attention (per b,h, 64-query tile) ----------------
// Q pre-scaled by 1/sqrt(dk). Block: 128 thr = 4 waves; wave owns 16 queries.
// K tile staged by TDM (overlapped with manual V-transpose staging).
__global__ __launch_bounds__(128) void flash_attn(
    const bf16* __restrict__ Qp, const bf16* __restrict__ Kp,
    const bf16* __restrict__ Vp, bf16* __restrict__ Xc)
{
    __shared__ bf16 Ks[64][72];        // K tile  [key][d] (TDM-padded pitch)
    __shared__ bf16 Vt[64][72];        // V tile transposed [d][key]
    __shared__ bf16 Ps[4][16][72];     // per-wave P tile [q][key]

    const int tid  = threadIdx.x;
    const int lane = tid & 31;
    const int wave = tid >> 5;
    const int half = lane >> 4;
    const int ln   = lane & 15;
    const int bh   = blockIdx.y;
    const int b    = bh >> 4;
    const int h    = bh & 15;
    const int qt   = blockIdx.x;
    const size_t rowbase = (size_t)b * kS;
    const int    colbase = h * kDK;
    const int    q0 = qt * 64 + wave * 16;

    const unsigned Ks_off = (unsigned)(uintptr_t)&Ks[0][0];

    // Q fragments (16 rows x 64 d) live in registers for the whole loop
    v16bf qa[2];
    for (int ks = 0; ks < 2; ++ks)
        qa[ks] = frag_nt(Qp + (rowbase + q0 + ln) * kD + colbase + ks * 32 + 8 * half);

    float mrun[8], lrun[8];
    v8f o[4] = {};
    for (int r = 0; r < 8; ++r) { mrun[r] = -3.0e38f; lrun[r] = 0.0f; }

    for (int kt = 0; kt < kS; kt += 64) {
        // async: TDM pulls the K tile while all threads transpose-stage V
        if (wave == 0)
            tdm_load_2d(Kp + (rowbase + kt) * kD + colbase, Ks_off, TDM_G1W0_P72,
                        (unsigned)kD, (unsigned)(kB * kS), 64u, 64u, (unsigned)kD);
        for (int i = 0; i < 4; ++i) {
            int c2  = tid * 4 + i;         // 0..511
            int row = c2 >> 3;             // key index 0..63
            int col = (c2 & 7) * 8;        // d offset
            v8bf vv = *reinterpret_cast<const v8bf*>(
                Vp + (rowbase + kt + row) * kD + colbase + col);
            for (int e = 0; e < 8; ++e) Vt[col + e][row] = vv[e];
        }
        if (wave == 0)
            __builtin_amdgcn_s_wait_tensorcnt(0);
        __syncthreads();

        // S = Q K^T   (4 accumulators of 16x16 over 64 keys)
        v8f s[4];
        for (int j = 0; j < 4; ++j) {
            v8f acc = {};
            for (int ks = 0; ks < 2; ++ks) {
                v16bf bfrag = frag_nt(&Ks[j * 16 + ln][ks * 32 + 8 * half]);
                acc = wmma_bf16(qa[ks], bfrag, acc);
            }
            s[j] = acc;
        }

        // online softmax per row (row lives in one VGPR across 16 lanes)
        for (int r = 0; r < 8; ++r) {
            float mx = s[0][r];
            for (int j = 1; j < 4; ++j) mx = fmaxf(mx, s[j][r]);
            for (int m = 1; m < 16; m <<= 1) mx = fmaxf(mx, __shfl_xor(mx, m, 32));
            float mnew = fmaxf(mrun[r], mx);
            float scl  = __expf(mrun[r] - mnew);
            float ps = 0.0f;
            for (int j = 0; j < 4; ++j) {
                float p = __expf(s[j][r] - mnew);
                s[j][r] = p;
                ps += p;
            }
            for (int m = 1; m < 16; m <<= 1) ps += __shfl_xor(ps, m, 32);
            lrun[r] = lrun[r] * scl + ps;
            mrun[r] = mnew;
            for (int j = 0; j < 4; ++j) o[j][r] *= scl;
        }

        // spill P (C layout) to LDS, reload as A fragments
        for (int j = 0; j < 4; ++j)
            for (int r = 0; r < 8; ++r)
                Ps[wave][r + 8 * half][j * 16 + ln] = (bf16)s[j][r];
        __syncthreads();

        v16bf pa[2];
        for (int ks = 0; ks < 2; ++ks)
            pa[ks] = frag_nt(&Ps[wave][ln][ks * 32 + 8 * half]);

        // O += P V   (B operand = Vt rows are output-d, K-contig over keys)
        for (int j = 0; j < 4; ++j)
            for (int ks = 0; ks < 2; ++ks) {
                v16bf vfrag = frag_nt(&Vt[j * 16 + ln][ks * 32 + 8 * half]);
                o[j] = wmma_bf16(pa[ks], vfrag, o[j]);
            }
        __syncthreads();
    }

    // normalize and write merged-head context [B*S, D] bf16
    for (int j = 0; j < 4; ++j)
        for (int r = 0; r < 8; ++r) {
            int qrow = q0 + r + 8 * half;
            float val = o[j][r] / lrun[r];
            Xc[(rowbase + qrow) * kD + colbase + j * 16 + ln] = (bf16)val;
        }
}

// ---------------- host-side orchestration ----------------
extern "C" void kernel_launch(void* const* d_in, const int* in_sizes, int n_in,
                              void* d_out, int out_size, void* d_ws, size_t ws_size,
                              hipStream_t stream) {
    (void)in_sizes; (void)n_in; (void)out_size; (void)ws_size;

    const float* q_in = (const float*)d_in[0];
    const float* k_in = (const float*)d_in[1];
    const float* v_in = (const float*)d_in[2];
    const float* Wq   = (const float*)d_in[3];
    const float* bq   = (const float*)d_in[4];
    const float* Wk   = (const float*)d_in[5];
    const float* bk   = (const float*)d_in[6];
    const float* Wv   = (const float*)d_in[7];
    const float* bv   = (const float*)d_in[8];
    const float* Wo   = (const float*)d_in[9];
    const float* bo   = (const float*)d_in[10];
    float* out = (float*)d_out;

    const size_t nIn = (size_t)kB * kS * kD;
    const size_t nW  = (size_t)kD * kD;

    bf16* base = (bf16*)d_ws;
    bf16* qinb = base;
    bf16* kinb = qinb + nIn;
    bf16* vinb = kinb + nIn;
    bf16* wqb  = vinb + nIn;
    bf16* wkb  = wqb + nW;
    bf16* wvb  = wkb + nW;
    bf16* wob  = wvb + nW;
    bf16* Qp   = wob + nW;
    bf16* Kp   = Qp + nIn;
    bf16* Vp   = Kp + nIn;
    bf16* Xc   = Vp + nIn;

    // 1) convert inputs + weights to bf16
    {
        int thr = 256;
        int blkIn = (int)((nIn / 8 + thr - 1) / thr);
        int blkW  = (int)((nW  / 8 + thr - 1) / thr);
        cvt_f32_bf16<<<blkIn, thr, 0, stream>>>(q_in, qinb, (int)nIn);
        cvt_f32_bf16<<<blkIn, thr, 0, stream>>>(k_in, kinb, (int)nIn);
        cvt_f32_bf16<<<blkIn, thr, 0, stream>>>(v_in, vinb, (int)nIn);
        cvt_f32_bf16<<<blkW,  thr, 0, stream>>>(Wq, wqb, (int)nW);
        cvt_f32_bf16<<<blkW,  thr, 0, stream>>>(Wk, wkb, (int)nW);
        cvt_f32_bf16<<<blkW,  thr, 0, stream>>>(Wv, wvb, (int)nW);
        cvt_f32_bf16<<<blkW,  thr, 0, stream>>>(Wo, wob, (int)nW);
    }

    // 2) projections (fold 1/sqrt(dk) into Q)
    const int M = kB * kS, N = kD, K = kD;
    dim3 ggrid(M / 64, N / 128);
    const float qscale = 0.125f;   // 1/sqrt(64)
    gemm_nt_bf16<<<ggrid, 256, 0, stream>>>(qinb, wqb, bq, Qp, nullptr, M, N, K, qscale);
    gemm_nt_bf16<<<ggrid, 256, 0, stream>>>(kinb, wkb, bk, Kp, nullptr, M, N, K, 1.0f);
    gemm_nt_bf16<<<ggrid, 256, 0, stream>>>(vinb, wvb, bv, Vp, nullptr, M, N, K, 1.0f);

    // 3) fused flash attention over all (b,h,q-tile)
    dim3 agrid(kS / 64, kB * kH);
    flash_attn<<<agrid, 128, 0, stream>>>(Qp, Kp, Vp, Xc);

    // 4) output projection -> f32 result
    gemm_nt_bf16<<<ggrid, 256, 0, stream>>>(Xc, wob, bo, nullptr, out, M, N, K, 1.0f);
}
